// Model_63196148793732
// MI455X (gfx1250) — matrix operand order repaired and checked by reading, not compiled, not run on hardware
//
#include <hip/hip_runtime.h>
#include <stdint.h>

typedef __bf16 bf16;
typedef __attribute__((ext_vector_type(16))) __bf16 v16bf;
typedef __attribute__((ext_vector_type(8)))  __bf16 v8bf;
typedef __attribute__((ext_vector_type(8)))  float  v8f;
typedef int v4i __attribute__((vector_size(16)));   // matches builtin param type

#define TB 256   // threads per block (8 waves, wave32)
#define BM 128   // block tile M
#define BN 128   // block tile N
#define BK 32    // K step (one bf16 WMMA K)
#define PK 40    // padded LDS K-stride in elements (80 bytes, 16B aligned)

#if defined(__gfx1250__) && __has_builtin(__builtin_amdgcn_global_load_async_to_lds_b128)
#define ASYNC_LDS 1
#else
#define ASYNC_LDS 0
#endif

#define AS1 __attribute__((address_space(1)))
#define AS3 __attribute__((address_space(3)))

// Wait for all of this wave's outstanding async (memory->LDS) transfers.
__device__ __forceinline__ void async_wait_all()
{
#if ASYNC_LDS
#if __has_builtin(__builtin_amdgcn_s_wait_asynccnt)
    __builtin_amdgcn_s_wait_asynccnt(0);
#else
    asm volatile("s_wait_asynccnt 0x0" ::: "memory");
#endif
#endif
}

// Move 16 bytes global -> LDS. Async DMA path on gfx1250 (no VGPR staging),
// synchronous load+ds_store fallback otherwise.
__device__ __forceinline__ void copy16_g2l(const bf16* g, bf16* l)
{
#if ASYNC_LDS
    __builtin_amdgcn_global_load_async_to_lds_b128(
        (AS1 v4i*)(uintptr_t)g,
        (AS3 v4i*)(uint32_t)(uintptr_t)l, 0, 0);
#else
    *(v8bf*)l = *(const v8bf*)g;
#endif
}

// ---------------------------------------------------------------------------
// TN bf16 GEMM: D[m][n] = scale * sum_k A[m][k]*B[n][k] (+ bias[m])
// A: [M][K] bf16, lda (K-contiguous).  B: [N][K] bf16, ldb (K-contiguous).
// D: [M][N], ldd; bf16 or f32 by template. Batched via blockIdx.z + strides.
// 8 waves as 4(M) x 2(N); wave tile 32x64 = 2x4 WMMA 16x16 tiles.
// Double-buffered LDS, async fill one K-stage ahead, one barrier per K-step.
// ---------------------------------------------------------------------------
template<bool OUT_BF16>
__global__ __launch_bounds__(TB)
void gemm_tn_bf16(const bf16* __restrict__ A, const bf16* __restrict__ Bmat,
                  void* __restrict__ Dp, const float* __restrict__ bias,
                  int K, int lda, int ldb, int ldd,
                  long bsA, long bsB, long bsD, float scale)
{
    __shared__ __align__(16) bf16 shA[2][BM * PK];
    __shared__ __align__(16) bf16 shB[2][BN * PK];

    const int z = blockIdx.z;
    const bf16* Ab = A    + (long)z * bsA;
    const bf16* Bb = Bmat + (long)z * bsB;

    const int tid  = threadIdx.x;
    const int lane = tid & 31;
    const int wave = tid >> 5;
    const int lm   = lane & 15;   // M (A) / N (B,D) index within tile
    const int h    = lane >> 4;   // lane-half selects K sub-range / M offset
    const int wm   = wave >> 1;   // 0..3
    const int wn   = wave & 1;    // 0..1
    const int m0   = blockIdx.y * BM;
    const int n0   = blockIdx.x * BN;
    const int wmo  = wm * 32;
    const int wno  = wn * 64;

    // Fill decomposition: 512 chunks of 16B per operand block.
    const int fr  = tid >> 2;         // rows 0..63 (+64 on 2nd pass)
    const int fko = (tid & 3) * 8;    // K offset in elements

    v8f acc[2][4] = {};

    // Prologue: fill stage 0.
    #pragma unroll
    for (int it = 0; it < 2; ++it) {
        int row = fr + it * 64;
        copy16_g2l(Ab + (long)(m0 + row) * lda + fko, &shA[0][row * PK + fko]);
        copy16_g2l(Bb + (long)(n0 + row) * ldb + fko, &shB[0][row * PK + fko]);
    }
    async_wait_all();
    __syncthreads();

    int pb = 0;
    for (int k0 = 0; k0 < K; k0 += BK) {
        const int nb = pb ^ 1;
        // Kick off next stage's fill (overlaps with WMMAs below).
        if (k0 + BK < K) {
            #pragma unroll
            for (int it = 0; it < 2; ++it) {
                int row = fr + it * 64;
                copy16_g2l(Ab + (long)(m0 + row) * lda + (k0 + BK) + fko,
                           &shA[nb][row * PK + fko]);
                copy16_g2l(Bb + (long)(n0 + row) * ldb + (k0 + BK) + fko,
                           &shB[nb][row * PK + fko]);
            }
        }

        // A fragment (16-bit A 16x32 layout): runs [8h,8h+8) and [16+8h,+8).
        v16bf af[2];
        #pragma unroll
        for (int i = 0; i < 2; ++i) {
            int r = wmo + i * 16 + lm;
            v8bf* pa = (v8bf*)&af[i];
            pa[0] = *(const v8bf*)(&shA[pb][r * PK + 8 * h]);
            pa[1] = *(const v8bf*)(&shA[pb][r * PK + 16 + 8 * h]);
        }
        // B fragment (16-bit B 32x16): runs [8h,8h+8) and [16+8h,+8).
        v16bf bfr[4];
        #pragma unroll
        for (int j = 0; j < 4; ++j) {
            int r = wno + j * 16 + lm;
            v8bf* pb2 = (v8bf*)&bfr[j];
            pb2[0] = *(const v8bf*)(&shB[pb][r * PK + 8 * h]);
            pb2[1] = *(const v8bf*)(&shB[pb][r * PK + 16 + 8 * h]);
        }

        #pragma unroll
        for (int i = 0; i < 2; ++i)
            #pragma unroll
            for (int j = 0; j < 4; ++j)
                acc[i][j] = __builtin_amdgcn_wmma_f32_16x16x32_bf16(
                    false, af[i], false, bfr[j],
                    (short)0, acc[i][j], false, false);

        // My next-stage transfers are done; barrier also guarantees everyone
        // finished reading buffer pb before it is refilled next iteration.
        async_wait_all();
        __syncthreads();
        pb = nb;
    }

    // Epilogue. C/D f32 16x16 layout: reg r holds row M = r + 8h, col N = lm.
    #pragma unroll
    for (int i = 0; i < 2; ++i) {
        #pragma unroll
        for (int j = 0; j < 4; ++j) {
            int tm = m0 + wmo + i * 16;
            int tn = n0 + wno + j * 16 + lm;
            #pragma unroll
            for (int r = 0; r < 8; ++r) {
                int row = tm + r + 8 * h;
                float v = acc[i][j][r] * scale;
                if (bias) v += bias[row];
                if (OUT_BF16)
                    ((bf16*)Dp)[(long)z * bsD + (long)row * ldd + tn] = (bf16)v;
                else
                    ((float*)Dp)[(long)z * bsD + (long)row * ldd + tn] = v;
            }
        }
    }
}

// ---------------------------------------------------------------------------
// Tiled transpose (+convert to bf16): src [rows][cols] -> dst [cols][rows].
// Batched via blockIdx.z (stride rows*cols both sides).
// ---------------------------------------------------------------------------
template<typename TS>
__global__ __launch_bounds__(256)
void transpose_to_bf16(const TS* __restrict__ src, bf16* __restrict__ dst,
                       int rows, int cols)
{
    __shared__ bf16 tile[32][33];
    const long boff = (long)blockIdx.z * rows * cols;
    const int r0 = blockIdx.y * 32;
    const int c0 = blockIdx.x * 32;
    const int tx = threadIdx.x & 31;
    const int ty = threadIdx.x >> 5;   // 0..7
    #pragma unroll
    for (int e = 0; e < 4; ++e) {
        int r = ty + e * 8;
        tile[r][tx] = (bf16)(float)src[boff + (long)(r0 + r) * cols + (c0 + tx)];
    }
    __syncthreads();
    #pragma unroll
    for (int e = 0; e < 4; ++e) {
        int r = ty + e * 8;
        dst[boff + (long)(c0 + r) * rows + (r0 + tx)] = tile[tx][r];
    }
}

__global__ void cvt_f32_to_bf16(const float* __restrict__ s,
                                bf16* __restrict__ d, int n)
{
    int i = blockIdx.x * blockDim.x + threadIdx.x;
    if (i < n) d[i] = (bf16)s[i];
}

// Row-wise softmax over 512 fp32 values -> bf16. One block per row.
__global__ __launch_bounds__(256)
void softmax512(const float* __restrict__ s, bf16* __restrict__ p)
{
    const long base = (long)blockIdx.x * 512;
    __shared__ float red[256];
    const int tid = threadIdx.x;
    float v0 = s[base + tid];
    float v1 = s[base + tid + 256];
    red[tid] = fmaxf(v0, v1);
    __syncthreads();
    for (int o = 128; o > 0; o >>= 1) {
        if (tid < o) red[tid] = fmaxf(red[tid], red[tid + o]);
        __syncthreads();
    }
    float m = red[0];
    __syncthreads();
    float e0 = __expf(v0 - m), e1 = __expf(v1 - m);
    red[tid] = e0 + e1;
    __syncthreads();
    for (int o = 128; o > 0; o >>= 1) {
        if (tid < o) red[tid] += red[tid + o];
        __syncthreads();
    }
    float inv = 1.0f / red[0];
    p[base + tid]       = (bf16)(e0 * inv);
    p[base + tid + 256] = (bf16)(e1 * inv);
}

// ---------------------------------------------------------------------------
extern "C" void kernel_launch(void* const* d_in, const int* in_sizes, int n_in,
                              void* d_out, int out_size, void* d_ws, size_t ws_size,
                              hipStream_t stream)
{
    (void)in_sizes; (void)n_in; (void)out_size; (void)ws_size;
    const int Bn = 8, Cc = 512, T = 4096, KC = 512, VC = 512;

    const float* x   = (const float*)d_in[0];
    const float* Wqk = (const float*)d_in[1];
    const float* bqk = (const float*)d_in[2];
    const float* Wv  = (const float*)d_in[3];
    const float* bv  = (const float*)d_in[4];

    char* ws = (char*)d_ws;
    size_t off = 0;
    bf16*  xT   = (bf16*)(ws + off); off += (size_t)Bn * T * Cc * 2;   // 32 MB
    bf16*  wqkb = (bf16*)(ws + off); off += (size_t)KC * Cc * 2;
    bf16*  wvb  = (bf16*)(ws + off); off += (size_t)VC * Cc * 2;
    bf16*  qk   = (bf16*)(ws + off); off += (size_t)Bn * KC * T * 2;   // 32 MB
    bf16*  val  = (bf16*)(ws + off); off += (size_t)Bn * VC * T * 2;   // 32 MB
    bf16*  valT = (bf16*)(ws + off); off += (size_t)Bn * T * VC * 2;   // 32 MB
    float* sc   = (float*)(ws + off); off += (size_t)Bn * KC * VC * 4; //  8 MB
    bf16*  p    = (bf16*)(ws + off); off += (size_t)Bn * KC * VC * 2;  //  4 MB

    // 1) Convert weights to bf16 (layout [out_ch][in_ch] is already K-major).
    cvt_f32_to_bf16<<<(KC * Cc + 255) / 256, 256, 0, stream>>>(Wqk, wqkb, KC * Cc);
    cvt_f32_to_bf16<<<(VC * Cc + 255) / 256, 256, 0, stream>>>(Wv,  wvb,  VC * Cc);

    // 2) x [B][C][T] fp32 -> xT [B][T][C] bf16.
    transpose_to_bf16<float><<<dim3(T / 32, Cc / 32, Bn), 256, 0, stream>>>(
        x, xT, Cc, T);

    // 3) Projections: qk/val[k][t] = W[k][:] . xT[t][:] + bias[k].
    gemm_tn_bf16<true><<<dim3(T / BN, KC / BM, Bn), TB, 0, stream>>>(
        wqkb, xT, qk, bqk, /*K=*/Cc, /*lda=*/Cc, /*ldb=*/Cc, /*ldd=*/T,
        /*bsA=*/0L, /*bsB=*/(long)T * Cc, /*bsD=*/(long)KC * T, 1.0f);
    gemm_tn_bf16<true><<<dim3(T / BN, VC / BM, Bn), TB, 0, stream>>>(
        wvb, xT, val, bv, Cc, Cc, Cc, T,
        0L, (long)T * Cc, (long)VC * T, 1.0f);

    // 4) val [B][V][T] -> valT [B][T][V] (bf16 transpose).
    transpose_to_bf16<bf16><<<dim3(T / 32, VC / 32, Bn), 256, 0, stream>>>(
        val, valT, VC, T);

    // 5) scores[k][v] = (1/512) * qk[k][:] . val[v][:]  (reduce over T).
    gemm_tn_bf16<false><<<dim3(VC / BN, KC / BM, Bn), TB, 0, stream>>>(
        qk, val, sc, nullptr, /*K=*/T, /*lda=*/T, /*ldb=*/T, /*ldd=*/VC,
        (long)KC * T, (long)VC * T, (long)KC * VC, 1.0f / 512.0f);

    // 6) softmax over v, fp32 -> bf16 p.
    softmax512<<<Bn * KC, 256, 0, stream>>>(sc, p);

    // 7) out[k][t] = p[k][:] . valT[t][:]  -> fp32 d_out [B][C][T].
    gemm_tn_bf16<false><<<dim3(T / BN, KC / BM, Bn), TB, 0, stream>>>(
        p, valT, (float*)d_out, nullptr, /*K=*/VC, /*lda=*/VC, /*ldb=*/VC,
        /*ldd=*/T, (long)KC * VC, (long)T * VC, (long)Cc * T, 1.0f);
}
